// VectorQuantizerEMA_50491635532272
// MI455X (gfx1250) — matrix operand order repaired and checked by reading, not compiled, not run on hardware
//
#include <hip/hip_runtime.h>
#include <hip/hip_bf16.h>

// ---------------------------------------------------------------------------
// VQ-VAE EMA quantizer for gfx1250 (MI455X).
//   z:          [32,128,64,64] f32   -> flat_z [N=131072, D=128], column d
//                                       contiguous in n (stride HW=4096 per d)
//   embedding:  [128,512] f32
// Distances via D = -2 * A(f16) x B(f16) with v_wmma_f32_16x16x32_f16,
// ||e||^2 folded in, ||z||^2 left out (argmin-invariant).
// Two code tiles in flight per iteration to fill WMMA->VALU hazard slots.
// ---------------------------------------------------------------------------

typedef __attribute__((ext_vector_type(16))) _Float16 v16h;
typedef __attribute__((ext_vector_type(8)))  _Float16 v8h;
typedef __attribute__((ext_vector_type(8)))  float    v8f;

#define NUM_K   512
#define DIMS    128
#define NROWS   131072          // 32*64*64
#define HW      4096            // 64*64
#define EPAD    136             // padded f16 row stride for codebook in LDS
#define QELEMS  16777216        // 32*128*64*64

// ---------------------------------------------------------------- init ws --
__global__ __launch_bounds__(256) void vq_init_kernel(float* p, int n) {
    int i = blockIdx.x * 256 + threadIdx.x;
    if (i < n) p[i] = 0.0f;
}

// ----------------------------------------------------- distance + argmin ---
// grid: 1024 blocks (128 rows each), 256 threads (8 waves, 16 rows/wave)
__global__ __launch_bounds__(256)
void vq_dist_kernel(const float* __restrict__ z,
                    const float* __restrict__ emb,
                    float* __restrict__ outIdx,   // [N] argmin as float
                    float* __restrict__ wsEnc)    // [K] one-hot counts
{
    __shared__ _Float16 sE[NUM_K * EPAD];   // f16 codebook, padded rows
    __shared__ float    sEn2[NUM_K];        // ||e_k||^2

    const int tid = threadIdx.x;

    // ---- stage full codebook into LDS as f16 (coalesced: k across threads)
    for (int k = tid; k < NUM_K; k += 256) {
        float s = 0.0f;
        #pragma unroll 4
        for (int d = 0; d < DIMS; ++d) {
            float v = emb[d * NUM_K + k];
            s += v * v;
            sE[k * EPAD + d] = (_Float16)v;
        }
        sEn2[k] = s;
    }
    __syncthreads();

    const int wave = tid >> 5;
    const int lane = tid & 31;
    const int half = lane >> 4;      // 0: lanes 0-15, 1: lanes 16-31
    const int r    = lane & 15;

    const int rowBase = blockIdx.x * 128 + wave * 16;
    const int row = rowBase + r;                 // this lane's A-matrix row
    const int b   = row >> 12;                   // row / 4096
    const int hw  = row & 4095;
    const float* __restrict__ zrow = z + (size_t)b * (DIMS * HW) + hw;
    __builtin_prefetch(zrow, 0, 1);              // global_prefetch_b8

    // ---- A fragments (16x32 f16 per the CDNA5 layout) + ||z||^2 partial
    v16h aF[4];
    float sq = 0.0f;
    #pragma unroll
    for (int c = 0; c < 4; ++c) {
        #pragma unroll
        for (int j = 0; j < 8; ++j) {
            int d0 = c * 32 +      half * 8 + j;   // K 0..7  / 8..15
            int d1 = c * 32 + 16 + half * 8 + j;   // K 16..23 / 24..31
            float f0 = zrow[(size_t)d0 * HW];
            float f1 = zrow[(size_t)d1 * HW];
            aF[c][j]     = (_Float16)f0;
            aF[c][8 + j] = (_Float16)f1;
            sq += f0 * f0 + f1 * f1;
        }
    }
    sq += __shfl_xor(sq, 16, 32);   // full ||z_row||^2 (both halves hold it)
    (void)sq;

    // ---- sweep 32 code tiles, 2 per iteration (independent WMMA chains)
    float bestS[8];
    int   bestI[8];
    #pragma unroll
    for (int v = 0; v < 8; ++v) { bestS[v] = 3.4e38f; bestI[v] = 0; }

    for (int t = 0; t < 32; t += 2) {
        const int k0 = t * 16 + r;                // codes t*16..t*16+15
        const int k1 = k0 + 16;                   // codes (t+1)*16..
        const _Float16* bp0 = &sE[k0 * EPAD + half * 8];
        const _Float16* bp1 = &sE[k1 * EPAD + half * 8];
        v8f acc0 = {};
        v8f acc1 = {};
        #pragma unroll
        for (int c = 0; c < 4; ++c) {
            v8h lo0 = *(const v8h*)(bp0 + c * 32);
            v8h hi0 = *(const v8h*)(bp0 + c * 32 + 16);
            v8h lo1 = *(const v8h*)(bp1 + c * 32);
            v8h hi1 = *(const v8h*)(bp1 + c * 32 + 16);
            v16h b0, b1;
            #pragma unroll
            for (int j = 0; j < 8; ++j) {
                b0[j] = lo0[j]; b0[8 + j] = hi0[j];
                b1[j] = lo1[j]; b1[8 + j] = hi1[j];
            }
            acc0 = __builtin_amdgcn_wmma_f32_16x16x32_f16(
                       false, aF[c], false, b0, (short)0, acc0, false, false);
            acc1 = __builtin_amdgcn_wmma_f32_16x16x32_f16(
                       false, aF[c], false, b1, (short)0, acc1, false, false);
        }
        const float en0 = sEn2[k0];
        const float en1 = sEn2[k1];
        #pragma unroll
        for (int v = 0; v < 8; ++v) {
            float s0 = fmaf(-2.0f, acc0[v], en0);  // argmin-equivalent score
            if (s0 < bestS[v]) { bestS[v] = s0; bestI[v] = k0; }
            float s1 = fmaf(-2.0f, acc1[v], en1);
            if (s1 < bestS[v]) { bestS[v] = s1; bestI[v] = k1; }
        }
    }

    // ---- reduce (dist,idx) across the 16 lanes holding the same row set
    #pragma unroll
    for (int m = 1; m < 16; m <<= 1) {
        #pragma unroll
        for (int v = 0; v < 8; ++v) {
            float os = __shfl_xor(bestS[v], m, 32);
            int   oi = __shfl_xor(bestI[v], m, 32);
            if (os < bestS[v] || (os == bestS[v] && oi < bestI[v])) {
                bestS[v] = os; bestI[v] = oi;
            }
        }
    }

    if (r == 0) {   // lanes 0 and 16 own rows M=0..7 and M=8..15 respectively
        #pragma unroll
        for (int v = 0; v < 8; ++v) {
            int M = 8 * half + v;
            int grow = rowBase + M;
            int ki = bestI[v];
            outIdx[grow] = (float)ki;
            atomicAdd(&wsEnc[ki], 1.0f);
        }
    }
}

// --------------------------------------- quantize + loss + embed_sum scatter
// grid: 128(d) x 32(image) = 4096 blocks, 256 threads, 16 n-iters each.
// LDS histogram per (block, d) then <=512 global atomics per block.
__global__ __launch_bounds__(256)
void vq_quant_kernel(const float* __restrict__ z,
                     const float* __restrict__ emb,
                     const float* __restrict__ idxF,   // [N]
                     float* __restrict__ outQ,
                     float* __restrict__ wsES,         // [D,K] embed_sum
                     float* __restrict__ wsLoss)
{
    __shared__ float sAcc[NUM_K];
    __shared__ float sL[8];

    const int d   = blockIdx.x >> 5;          // 0..127
    const int img = blockIdx.x & 31;          // batch index b (HW = 4096)
    const int tid = threadIdx.x;

    for (int i = tid; i < NUM_K; i += 256) sAcc[i] = 0.0f;
    __syncthreads();

    const float* __restrict__ erow = emb + d * NUM_K;
    const size_t base  = (size_t)img * (DIMS * HW) + (size_t)d * HW;
    const int    nBase = img * HW;

    float lsum = 0.0f;
    #pragma unroll 4
    for (int it = 0; it < 16; ++it) {
        const int hw = it * 256 + tid;
        const int k  = (int)idxF[nBase + hw];
        const float zv = z[base + hw];
        const float ev = erow[k];
        outQ[base + hw] = ev;                 // straight-through forward value
        atomicAdd(&sAcc[k], zv);              // DS atomic: partial embed_sum
        const float df = ev - zv;
        lsum += df * df;                      // exact f32 loss term
    }

    // loss: wave butterfly + cross-wave LDS reduce
    #pragma unroll
    for (int m = 16; m >= 1; m >>= 1) lsum += __shfl_xor(lsum, m, 32);
    if ((tid & 31) == 0) sL[tid >> 5] = lsum;
    __syncthreads();
    if (tid == 0) {
        float t = 0.0f;
        #pragma unroll
        for (int w = 0; w < 8; ++w) t += sL[w];
        atomicAdd(wsLoss, t);
    }

    // push per-block partial sums (skip empty bins)
    for (int i = tid; i < NUM_K; i += 256) {
        const float v = sAcc[i];
        if (v != 0.0f) atomicAdd(&wsES[d * NUM_K + i], v);
    }
}

// ----------------------------------------------------------- EMA finalize --
// single block of 512 threads (one per code)
__global__ __launch_bounds__(512)
void vq_final_kernel(const float* __restrict__ clusterSize,
                     const float* __restrict__ embedAvg,
                     const float* __restrict__ wsEnc,
                     const float* __restrict__ wsES,
                     const float* __restrict__ wsLoss,
                     float* __restrict__ outLoss,
                     float* __restrict__ outNewEmb,
                     float* __restrict__ outNewCS,
                     float* __restrict__ outNewEA)
{
    __shared__ float sRed[NUM_K];
    const int k = threadIdx.x;

    float ncs = clusterSize[k] * 0.99f + 0.01f * wsEnc[k];
    outNewCS[k] = ncs;
    sRed[k] = ncs;
    __syncthreads();
    for (int s = 256; s > 0; s >>= 1) {
        if (k < s) sRed[k] += sRed[k + s];
        __syncthreads();
    }
    const float nTot = sRed[0];
    const float nmax = fmaxf(nTot, 1e-5f);
    const float csk  = (ncs + 1e-5f) / (nmax + 512.0f * 1e-5f) * nmax;

    #pragma unroll 4
    for (int d = 0; d < DIMS; ++d) {
        int o = d * NUM_K + k;
        float nea = embedAvg[o] * 0.99f + 0.01f * wsES[o];
        outNewEA[o]  = nea;
        outNewEmb[o] = nea / csk;
    }
    if (k == 0) outLoss[0] = 1.25f * wsLoss[0] * (1.0f / (float)QELEMS);
}

// ---------------------------------------------------------------------------
extern "C" void kernel_launch(void* const* d_in, const int* in_sizes, int n_in,
                              void* d_out, int out_size, void* d_ws, size_t ws_size,
                              hipStream_t stream) {
    const float* z   = (const float*)d_in[0];   // 16777216
    const float* emb = (const float*)d_in[1];   // 65536
    const float* cs  = (const float*)d_in[2];   // 512
    const float* ea  = (const float*)d_in[3];   // 65536

    float* out     = (float*)d_out;
    float* outQ    = out;                                  // 16777216
    float* outLoss = out + QELEMS;                         // 1
    float* outIdx  = out + QELEMS + 1;                     // 131072
    float* outNE   = outIdx + NROWS;                       // 65536
    float* outNCS  = outNE + DIMS * NUM_K;                 // 512
    float* outNEA  = outNCS + NUM_K;                       // 65536

    float* wsES   = (float*)d_ws;                          // 65536
    float* wsEnc  = wsES + DIMS * NUM_K;                   // 512
    float* wsLoss = wsEnc + NUM_K;                         // 1
    const int nInit = DIMS * NUM_K + NUM_K + 1;            // 66049

    hipLaunchKernelGGL(vq_init_kernel, dim3((nInit + 255) / 256), dim3(256), 0,
                       stream, wsES, nInit);
    hipLaunchKernelGGL(vq_dist_kernel, dim3(NROWS / 128), dim3(256), 0, stream,
                       z, emb, outIdx, wsEnc);
    hipLaunchKernelGGL(vq_quant_kernel, dim3(128 * 32), dim3(256), 0, stream,
                       z, emb, outIdx, outQ, wsES, wsLoss);
    hipLaunchKernelGGL(vq_final_kernel, dim3(1), dim3(512), 0, stream,
                       cs, ea, wsEnc, wsES, wsLoss,
                       outLoss, outNE, outNCS, outNEA);
}
